// PointNetPPCls_40303973106263
// MI455X (gfx1250) — matrix operand order repaired and checked by reading, not compiled, not run on hardware
//
#include <hip/hip_runtime.h>
#include <hip/hip_bf16.h>
#include <math.h>

typedef _Float16 half_t;
typedef __attribute__((ext_vector_type(16))) _Float16 v16h;
typedef __attribute__((ext_vector_type(8)))  _Float16 v8h;
typedef __attribute__((ext_vector_type(8)))  float    v8f;

#define EPSV 1e-5f

// ---------------------------------------------------------------------------
// Layout transpose: xyz (B,3,N) -> pts (B,N,3)
// ---------------------------------------------------------------------------
__global__ void k_transpose_xyz(const float* __restrict__ xyz, float* __restrict__ pts,
                                int Bv, int N) {
  int i = blockIdx.x * blockDim.x + threadIdx.x;
  int total = Bv * N * 3;
  if (i >= total) return;
  int d = i % 3;
  int n = (i / 3) % N;
  int b = i / (3 * N);
  pts[i] = xyz[((size_t)b * 3 + d) * N + n];
}

// ---------------------------------------------------------------------------
// Farthest point sampling: one block per batch, dist kept in LDS.
// Matches jnp.argmax tie-break (first/lowest index).
// ---------------------------------------------------------------------------
__global__ __launch_bounds__(256) void k_fps(const float* __restrict__ pts, int N, int npoint,
                                             int* __restrict__ cent) {
  __shared__ float dist[4096];
  __shared__ float rv[256];
  __shared__ int   ri[256];
  __shared__ float cx, cy, cz;
  __shared__ int   s_far;
  int b = blockIdx.x;
  int tid = threadIdx.x;
  const float* P = pts + (size_t)b * N * 3;
  for (int n = tid; n < N; n += 256) dist[n] = 1e10f;
  if (tid == 0) s_far = 0;
  __syncthreads();
  for (int i = 0; i < npoint; i++) {
    if (tid == 0) {
      int far = s_far;
      cent[b * npoint + i] = far;
      cx = P[far * 3 + 0];
      cy = P[far * 3 + 1];
      cz = P[far * 3 + 2];
    }
    __syncthreads();
    float bv = -1e30f; int bi = N;
    for (int n = tid; n < N; n += 256) {
      float dx = P[n * 3 + 0] - cx;
      float dy = P[n * 3 + 1] - cy;
      float dz = P[n * 3 + 2] - cz;
      float d = dx * dx + dy * dy + dz * dz;
      float old = dist[n];
      d = fminf(old, d);
      dist[n] = d;
      if (d > bv || (d == bv && n < bi)) { bv = d; bi = n; }
    }
    rv[tid] = bv; ri[tid] = bi;
    __syncthreads();
    for (int s = 128; s > 0; s >>= 1) {
      if (tid < s) {
        float ov = rv[tid + s]; int oi = ri[tid + s];
        if (ov > rv[tid] || (ov == rv[tid] && oi < ri[tid])) { rv[tid] = ov; ri[tid] = oi; }
      }
      __syncthreads();
    }
    if (tid == 0) s_far = ri[0];
    __syncthreads();
  }
}

// newxyz[b][m][:] = pts[b][cent[b][m]][:]
__global__ void k_gather_xyz(const float* __restrict__ pts, const int* __restrict__ cent,
                             float* __restrict__ newxyz, int Bv, int N, int M) {
  int i = blockIdx.x * blockDim.x + threadIdx.x;
  if (i >= Bv * M) return;
  int b = i / M;
  int n = cent[i];
  newxyz[i * 3 + 0] = pts[((size_t)b * N + n) * 3 + 0];
  newxyz[i * 3 + 1] = pts[((size_t)b * N + n) * 3 + 1];
  newxyz[i * 3 + 2] = pts[((size_t)b * N + n) * 3 + 2];
}

// ball query: first nsample in-radius indices in ascending order, tail filled with first
__global__ void k_ball_query(const float* __restrict__ pts, const float* __restrict__ newxyz,
                             int Bv, int N, int M, int nsample, float r2,
                             int* __restrict__ idx) {
  int i = blockIdx.x * blockDim.x + threadIdx.x;
  if (i >= Bv * M) return;
  int b = i / M;
  const float* P = pts + (size_t)b * N * 3;
  float qx = newxyz[i * 3 + 0];
  float qy = newxyz[i * 3 + 1];
  float qz = newxyz[i * 3 + 2];
  int* out = idx + (size_t)i * nsample;
  int cnt = 0;
  for (int n = 0; n < N && cnt < nsample; n++) {
    float dx = P[n * 3 + 0] - qx;
    float dy = P[n * 3 + 1] - qy;
    float dz = P[n * 3 + 2] - qz;
    float d = dx * dx + dy * dy + dz * dz;
    if (d <= r2) out[cnt++] = n;
  }
  int first = (cnt > 0) ? out[0] : 0;
  for (; cnt < nsample; cnt++) out[cnt] = first;
}

// grouped xyz only (SA1): rows = B*M*S, cols [0..2]=rel xyz, rest zero-pad
__global__ void k_group_xyz(const float* __restrict__ pts, const float* __restrict__ newxyz,
                            const int* __restrict__ idx, half_t* __restrict__ Xh,
                            int Bv, int N, int M, int S, int Kp) {
  int i = blockIdx.x * blockDim.x + threadIdx.x;
  if (i >= Bv * M * S) return;
  int m = (i / S) % M;
  int b = i / (S * M);
  int n = idx[i];
  half_t* dst = Xh + (size_t)i * Kp;
  for (int d = 0; d < 3; d++)
    dst[d] = (half_t)(pts[((size_t)b * N + n) * 3 + d] - newxyz[((size_t)b * M + m) * 3 + d]);
  for (int k = 3; k < Kp; k++) dst[k] = (half_t)0.f;
}

// grouped xyz + feature concat (SA2)
__global__ void k_group_feat(const float* __restrict__ pts, const float* __restrict__ newxyz,
                             const int* __restrict__ idx, const float* __restrict__ feat,
                             half_t* __restrict__ Xh,
                             int Bv, int N, int M, int S, int C, int Kp) {
  int i = blockIdx.x * blockDim.x + threadIdx.x;
  if (i >= Bv * M * S) return;
  int m = (i / S) % M;
  int b = i / (S * M);
  int n = idx[i];
  half_t* dst = Xh + (size_t)i * Kp;
  for (int d = 0; d < 3; d++)
    dst[d] = (half_t)(pts[((size_t)b * N + n) * 3 + d] - newxyz[((size_t)b * M + m) * 3 + d]);
  const float* fp = feat + ((size_t)b * N + n) * C;
  for (int c = 0; c < C; c++) dst[3 + c] = (half_t)fp[c];
  for (int k = 3 + C; k < Kp; k++) dst[k] = (half_t)0.f;
}

// group_all (SA3): feats = concat(xyz, points), no centering
__global__ void k_group_all(const float* __restrict__ pxyz, const float* __restrict__ feat,
                            half_t* __restrict__ Xh, int Bv, int S, int C, int Kp) {
  int i = blockIdx.x * blockDim.x + threadIdx.x;
  if (i >= Bv * S) return;
  half_t* dst = Xh + (size_t)i * Kp;
  for (int d = 0; d < 3; d++) dst[d] = (half_t)pxyz[(size_t)i * 3 + d];
  const float* fp = feat + (size_t)i * C;
  for (int c = 0; c < C; c++) dst[3 + c] = (half_t)fp[c];
  for (int k = 3 + C; k < Kp; k++) dst[k] = (half_t)0.f;
}

// weight f32 (Cout x Cin) -> f16 (Cout x Kp) zero-padded
__global__ void k_wconv(const float* __restrict__ W, half_t* __restrict__ Wh,
                        int Cout, int Cin, int Kp) {
  int i = blockIdx.x * blockDim.x + threadIdx.x;
  if (i >= Cout * Kp) return;
  int k = i % Kp;
  int o = i / Kp;
  Wh[i] = (k < Cin) ? (half_t)W[(size_t)o * Cin + k] : (half_t)0.f;
}

__global__ void k_f2h(const float* __restrict__ X, half_t* __restrict__ Xh, size_t total) {
  size_t i = (size_t)blockIdx.x * blockDim.x + threadIdx.x;
  if (i >= total) return;
  Xh[i] = (half_t)X[i];
}

// ---------------------------------------------------------------------------
// WMMA fragment loaders (fold into paired global_load_b128 — verified round 1).
// A(16x32 f16): lane m=lane&15; halves 0..15 = K 16*hh .. 16*hh+15 (contiguous)
// B(32x16 f16): lane n=lane&15; halves 0..15 = K 16*hh .. 16*hh+15 (contiguous)
// ---------------------------------------------------------------------------
__device__ __forceinline__ v16h ld_frag_a(const half_t* p) {
  v8h a0 = *(const v8h*)(p);
  v8h a1 = *(const v8h*)(p + 8);
  v16h a;
#pragma unroll
  for (int q = 0; q < 8; q++) { a[q] = a0[q]; a[q + 8] = a1[q]; }
  return a;
}
__device__ __forceinline__ v16h ld_frag_b(const half_t* p) {
  v8h b0 = *(const v8h*)(p);
  v8h b1 = *(const v8h*)(p + 8);
  v16h b;
#pragma unroll
  for (int q = 0; q < 8; q++) { b[q] = b0[q]; b[q + 8] = b1[q]; }
  return b;
}

// ---------------------------------------------------------------------------
// WMMA GEMM, 4 N-tiles per wave (Nc % 64 == 0, (R/16) % 8 == 0).
// 256-thread block = 8 waves; each wave computes 16 rows x 64 cols with 4
// named accumulators — branch-free inner loop so the A fragment is loaded
// once and reused 4x per K-step with no accumulator shuffling.
// ---------------------------------------------------------------------------
__global__ __launch_bounds__(256) void k_gemm_wmma4(const half_t* __restrict__ Xh,
                                                    const half_t* __restrict__ Wh,
                                                    const float* __restrict__ bias,
                                                    float* __restrict__ Y,
                                                    int R, int Kp, int Nc) {
  int lane = threadIdx.x & 31;
  int wave = threadIdx.x >> 5;            // 0..7
  int hh = lane >> 4;
  int l16 = lane & 15;
  int rt = blockIdx.y * 8 + wave;         // row tile
  int n0 = blockIdx.x * 64;               // first output column of this wave

  const half_t* xrow = Xh + (size_t)(rt * 16 + l16) * Kp + 16 * hh;
  const half_t* w0 = Wh + (size_t)(n0 + l16) * Kp + 16 * hh;
  const half_t* w1 = w0 + (size_t)16 * Kp;
  const half_t* w2 = w0 + (size_t)32 * Kp;
  const half_t* w3 = w0 + (size_t)48 * Kp;

  v8f acc0 = {}, acc1 = {}, acc2 = {}, acc3 = {};
  for (int k0 = 0; k0 < Kp; k0 += 32) {
    v16h a = ld_frag_a(xrow + k0);
    acc0 = __builtin_amdgcn_wmma_f32_16x16x32_f16(false, a, false, ld_frag_b(w0 + k0),
                                                  (short)0, acc0, false, false);
    acc1 = __builtin_amdgcn_wmma_f32_16x16x32_f16(false, a, false, ld_frag_b(w1 + k0),
                                                  (short)0, acc1, false, false);
    acc2 = __builtin_amdgcn_wmma_f32_16x16x32_f16(false, a, false, ld_frag_b(w2 + k0),
                                                  (short)0, acc2, false, false);
    acc3 = __builtin_amdgcn_wmma_f32_16x16x32_f16(false, a, false, ld_frag_b(w3 + k0),
                                                  (short)0, acc3, false, false);
  }

  float* yrow = Y + (size_t)(rt * 16 + 8 * hh) * Nc + n0 + l16;
  float b0 = bias ? bias[n0 + l16 + 0]  : 0.f;
  float b1 = bias ? bias[n0 + l16 + 16] : 0.f;
  float b2 = bias ? bias[n0 + l16 + 32] : 0.f;
  float b3 = bias ? bias[n0 + l16 + 48] : 0.f;
#pragma unroll
  for (int j = 0; j < 8; j++) {
    float* yr = yrow + (size_t)j * Nc;    // C/D: VGPR j -> row j + 8*hh
    yr[0]  = acc0[j] + b0;
    yr[16] = acc1[j] + b1;
    yr[32] = acc2[j] + b2;
    yr[48] = acc3[j] + b3;
  }
}

// ---------------------------------------------------------------------------
// WMMA GEMM, single 16x16 tile per wave (any Nc % 16 == 0) — used for Nc=80.
// ---------------------------------------------------------------------------
__global__ __launch_bounds__(32) void k_gemm_wmma1(const half_t* __restrict__ Xh,
                                                   const half_t* __restrict__ Wh,
                                                   const float* __restrict__ bias,
                                                   float* __restrict__ Y,
                                                   int R, int Kp, int Nc) {
  int lane = threadIdx.x & 31;
  int nt = blockIdx.x;
  int rt = blockIdx.y;
  int hh = lane >> 4;
  int l16 = lane & 15;
  const half_t* xrow = Xh + (size_t)(rt * 16 + l16) * Kp + 16 * hh;
  const half_t* wrow = Wh + (size_t)(nt * 16 + l16) * Kp + 16 * hh;
  v8f acc = {};
  for (int k0 = 0; k0 < Kp; k0 += 32) {
    v16h a = ld_frag_a(xrow + k0);
    v16h bm = ld_frag_b(wrow + k0);
    acc = __builtin_amdgcn_wmma_f32_16x16x32_f16(false, a, false, bm, (short)0, acc,
                                                 false, false);
  }
  int n = nt * 16 + l16;
  float bv = bias ? bias[n] : 0.0f;
#pragma unroll
  for (int j = 0; j < 8; j++) {
    int r = rt * 16 + j + 8 * hh;
    Y[(size_t)r * Nc + n] = acc[j] + bv;
  }
}

// per-column sums for BN (deterministic block reduce, one block per channel)
__global__ __launch_bounds__(256) void k_colstats(const float* __restrict__ Y, int R, int Nc,
                                                  float* __restrict__ sums,
                                                  float* __restrict__ sqs) {
  __shared__ float ss[256], sq[256];
  int n = blockIdx.x;
  int tid = threadIdx.x;
  float s = 0.f, q = 0.f;
  for (int r = tid; r < R; r += 256) {
    float v = Y[(size_t)r * Nc + n];
    s += v; q += v * v;
  }
  ss[tid] = s; sq[tid] = q;
  __syncthreads();
  for (int st = 128; st > 0; st >>= 1) {
    if (tid < st) { ss[tid] += ss[tid + st]; sq[tid] += sq[tid + st]; }
    __syncthreads();
  }
  if (tid == 0) { sums[n] = ss[0]; sqs[n] = sq[0]; }
}

// BN (batch stats over all rows) + ReLU; in-place f32, optional f16 copy for next GEMM
__global__ void k_bn_relu(float* __restrict__ Y, const float* __restrict__ sums,
                          const float* __restrict__ sqs, const float* __restrict__ g,
                          const float* __restrict__ be, half_t* __restrict__ outh,
                          int R, int Nc) {
  size_t i = (size_t)blockIdx.x * blockDim.x + threadIdx.x;
  size_t total = (size_t)R * Nc;
  if (i >= total) return;
  int n = (int)(i % Nc);
  float m = sums[n] / (float)R;
  float v = sqs[n] / (float)R - m * m;
  float x = Y[i];
  float y = g[n] * (x - m) * rsqrtf(v + EPSV) + be[n];
  y = fmaxf(y, 0.f);
  Y[i] = y;
  if (outh) outh[i] = (half_t)y;
}

// max over sample axis: in (G x S x Nc) -> out (G x Nc)
__global__ void k_maxpool(const float* __restrict__ Y, float* __restrict__ out,
                          int G, int S, int Nc) {
  int i = blockIdx.x * blockDim.x + threadIdx.x;
  if (i >= G * Nc) return;
  int n = i % Nc;
  int g = i / Nc;
  const float* p = Y + (size_t)g * S * Nc + n;
  float m = p[0];
  for (int s = 1; s < S; s++) m = fmaxf(m, p[(size_t)s * Nc]);
  out[i] = m;
}

// ---------------------------------------------------------------------------
// Self-attention small kernels (f32)
// ---------------------------------------------------------------------------
__global__ void k_attscore(const float* __restrict__ q, const float* __restrict__ k,
                           float* __restrict__ att, int Bv, int P, int C) {
  int i = blockIdx.x * blockDim.x + threadIdx.x;
  if (i >= Bv * P * P) return;
  int j = i % P;
  int r = (i / P) % P;
  int b = i / (P * P);
  const float* qp = q + ((size_t)b * P + r) * C;
  const float* kp = k + ((size_t)b * P + j) * C;
  float s = 0.f;
  for (int c = 0; c < C; c++) s += qp[c] * kp[c];
  att[i] = s;
}

__global__ void k_softmax_rows(float* __restrict__ att, int rows, int C) {
  int r = blockIdx.x * blockDim.x + threadIdx.x;
  if (r >= rows) return;
  float* p = att + (size_t)r * C;
  float mx = p[0];
  for (int c = 1; c < C; c++) mx = fmaxf(mx, p[c]);
  float s = 0.f;
  for (int c = 0; c < C; c++) { float e = expf(p[c] - mx); p[c] = e; s += e; }
  float inv = 1.f / s;
  for (int c = 0; c < C; c++) p[c] *= inv;
}

__global__ void k_attout(const float* __restrict__ att, const float* __restrict__ v,
                         const float* __restrict__ xin, const float* __restrict__ gamma,
                         float* __restrict__ out, int Bv, int P, int C) {
  int i = blockIdx.x * blockDim.x + threadIdx.x;
  if (i >= Bv * P * C) return;
  int c = i % C;
  int r = (i / C) % P;
  int b = i / (C * P);
  const float* ap = att + ((size_t)b * P + r) * P;
  const float* vp = v + (size_t)b * P * C + c;
  float s = 0.f;
  for (int j = 0; j < P; j++) s += ap[j] * vp[(size_t)j * C];
  out[i] = gamma[0] * s + xin[i];
}

// ---------------------------------------------------------------------------
// Head / channel-attention small kernels (f32, tiny R=8)
// ---------------------------------------------------------------------------
__global__ void k_linear(const float* __restrict__ X, const float* __restrict__ W,
                         const float* __restrict__ bias, float* __restrict__ Y,
                         int R, int Cin, int Cout, int dorelu) {
  int i = blockIdx.x * blockDim.x + threadIdx.x;
  if (i >= R * Cout) return;
  int o = i % Cout;
  int r = i / Cout;
  const float* xp = X + (size_t)r * Cin;
  const float* wp = W + (size_t)o * Cin;
  float s = bias ? bias[o] : 0.f;
  for (int c = 0; c < Cin; c++) s += xp[c] * wp[c];
  if (dorelu) s = fmaxf(s, 0.f);
  Y[i] = s;
}

__global__ void k_sig_res(float* __restrict__ x, const float* __restrict__ a, int total) {
  int i = blockIdx.x * blockDim.x + threadIdx.x;
  if (i >= total) return;
  x[i] = x[i] + 1.f / (1.f + expf(-2.f * a[i]));
}

__global__ void k_bn1d_relu(float* __restrict__ X, const float* __restrict__ g,
                            const float* __restrict__ be, int R, int C) {
  int c = blockIdx.x * blockDim.x + threadIdx.x;
  if (c >= C) return;
  float m = 0.f;
  for (int r = 0; r < R; r++) m += X[(size_t)r * C + c];
  m /= (float)R;
  float v = 0.f;
  for (int r = 0; r < R; r++) { float d = X[(size_t)r * C + c] - m; v += d * d; }
  v /= (float)R;
  float inv = rsqrtf(v + EPSV);
  for (int r = 0; r < R; r++)
    X[(size_t)r * C + c] = fmaxf(g[c] * (X[(size_t)r * C + c] - m) * inv + be[c], 0.f);
}

__global__ void k_concat_feat(const float* __restrict__ l3, const float* __restrict__ l2f,
                              float* __restrict__ x, int Bv) {
  int i = blockIdx.x * blockDim.x + threadIdx.x;
  if (i >= Bv * 1664) return;
  int c = i % 1664;
  int b = i / 1664;
  x[i] = (c < 1024) ? l3[b * 1024 + c] : l2f[b * 640 + (c - 1024)];
}

__global__ void k_logsoftmax2(const float* __restrict__ logits, float* __restrict__ out, int Bv) {
  int b = blockIdx.x * blockDim.x + threadIdx.x;
  if (b >= Bv) return;
  float a = logits[b * 2 + 0];
  float c = logits[b * 2 + 1];
  float mx = fmaxf(a, c);
  float ls = mx + logf(expf(a - mx) + expf(c - mx));
  out[b * 2 + 0] = a - ls;
  out[b * 2 + 1] = c - ls;
}

// ---------------------------------------------------------------------------
extern "C" void kernel_launch(void* const* d_in, const int* in_sizes, int n_in,
                              void* d_out, int out_size, void* d_ws, size_t ws_size,
                              hipStream_t stream) {
  (void)in_sizes; (void)n_in; (void)out_size; (void)ws_size;

  constexpr int Bv = 8, N1 = 4096, M1 = 512, S1 = 32, M2 = 128, S2 = 64, P = 128;
  constexpr int R1 = Bv * M1 * S1;   // 131072
  constexpr int R2 = Bv * M2 * S2;   // 65536
  constexpr int R3 = Bv * P;         // 1024

  // ---- input unpack (pytree leaf order) ----
  const float* xyz = (const float*)d_in[0];
  const float* sa1W[3]  = {(const float*)d_in[1], (const float*)d_in[5], (const float*)d_in[9]};
  const float* sa1b[3]  = {(const float*)d_in[2], (const float*)d_in[6], (const float*)d_in[10]};
  const float* sa1g[3]  = {(const float*)d_in[3], (const float*)d_in[7], (const float*)d_in[11]};
  const float* sa1be[3] = {(const float*)d_in[4], (const float*)d_in[8], (const float*)d_in[12]};
  const float* sa2W[3]  = {(const float*)d_in[13], (const float*)d_in[17], (const float*)d_in[21]};
  const float* sa2b[3]  = {(const float*)d_in[14], (const float*)d_in[18], (const float*)d_in[22]};
  const float* sa2g[3]  = {(const float*)d_in[15], (const float*)d_in[19], (const float*)d_in[23]};
  const float* sa2be[3] = {(const float*)d_in[16], (const float*)d_in[20], (const float*)d_in[24]};
  const float* sa3W[3]  = {(const float*)d_in[25], (const float*)d_in[29], (const float*)d_in[33]};
  const float* sa3b[3]  = {(const float*)d_in[26], (const float*)d_in[30], (const float*)d_in[34]};
  const float* sa3g[3]  = {(const float*)d_in[27], (const float*)d_in[31], (const float*)d_in[35]};
  const float* sa3be[3] = {(const float*)d_in[28], (const float*)d_in[32], (const float*)d_in[36]};
  const float* qW = (const float*)d_in[37]; const float* qb = (const float*)d_in[38];
  const float* kW = (const float*)d_in[39]; const float* kb = (const float*)d_in[40];
  const float* vW = (const float*)d_in[41]; const float* vb = (const float*)d_in[42];
  const float* gamma = (const float*)d_in[43];
  const float* caW1 = (const float*)d_in[44]; const float* caW2 = (const float*)d_in[45];
  const float* f1W = (const float*)d_in[46]; const float* f1b = (const float*)d_in[47];
  const float* b1g = (const float*)d_in[48]; const float* b1b = (const float*)d_in[49];
  const float* f2W = (const float*)d_in[50]; const float* f2b = (const float*)d_in[51];
  const float* b2g = (const float*)d_in[52]; const float* b2b = (const float*)d_in[53];
  const float* f3W = (const float*)d_in[54]; const float* f3b = (const float*)d_in[55];

  // ---- workspace carve-out ----
  char* wsb = (char*)d_ws;
  size_t off = 0;
  auto alloc = [&](size_t bytes) -> void* {
    void* p = wsb + off;
    off = (off + bytes + 255) & ~(size_t)255;
    return p;
  };
  float* pts1    = (float*)alloc((size_t)Bv * N1 * 3 * 4);
  int*   cent1   = (int*)  alloc((size_t)Bv * M1 * 4);
  float* newxyz1 = (float*)alloc((size_t)Bv * M1 * 3 * 4);
  int*   idx1    = (int*)  alloc((size_t)Bv * M1 * S1 * 4);
  int*   cent2   = (int*)  alloc((size_t)Bv * M2 * 4);
  float* newxyz2 = (float*)alloc((size_t)Bv * M2 * 3 * 4);
  int*   idx2    = (int*)  alloc((size_t)Bv * M2 * S2 * 4);
  half_t* HA     = (half_t*)alloc((size_t)R2 * 352 * 2);       // 46.1 MB (max f16 buf A)
  half_t* HB     = (half_t*)alloc((size_t)R1 * 64 * 2);        // 16.8 MB (max f16 buf B)
  float* Y       = (float*)alloc((size_t)R1 * 320 * 4);        // 167.8 MB gemm output
  float* sums    = (float*)alloc(1024 * 4);
  float* sqs     = (float*)alloc(1024 * 4);
  float* l1_p    = (float*)alloc((size_t)Bv * M1 * 320 * 4);
  float* l2_p    = (float*)alloc((size_t)Bv * M2 * 640 * 4);
  float* qf      = (float*)alloc((size_t)R3 * 80 * 4);
  float* kf      = (float*)alloc((size_t)R3 * 80 * 4);
  float* vf      = (float*)alloc((size_t)R3 * 640 * 4);
  float* att     = (float*)alloc((size_t)Bv * P * P * 4);
  float* l2att   = (float*)alloc((size_t)Bv * P * 640 * 4);
  float* l2feat  = (float*)alloc((size_t)Bv * 640 * 4);
  float* xcat    = (float*)alloc((size_t)Bv * 1664 * 4);
  float* a1      = (float*)alloc((size_t)Bv * 104 * 4);
  float* a2      = (float*)alloc((size_t)Bv * 1664 * 4);
  float* h1      = (float*)alloc((size_t)Bv * 512 * 4);
  float* h2      = (float*)alloc((size_t)Bv * 256 * 4);
  float* logits  = (float*)alloc((size_t)Bv * 2 * 4);
  half_t* w10 = (half_t*)alloc((size_t)64 * 32 * 2);
  half_t* w11 = (half_t*)alloc((size_t)64 * 64 * 2);
  half_t* w12 = (half_t*)alloc((size_t)320 * 64 * 2);
  half_t* w20 = (half_t*)alloc((size_t)128 * 352 * 2);
  half_t* w21 = (half_t*)alloc((size_t)128 * 128 * 2);
  half_t* w22 = (half_t*)alloc((size_t)640 * 128 * 2);
  half_t* w30 = (half_t*)alloc((size_t)256 * 672 * 2);
  half_t* w31 = (half_t*)alloc((size_t)512 * 256 * 2);
  half_t* w32 = (half_t*)alloc((size_t)1024 * 512 * 2);
  half_t* whq = (half_t*)alloc((size_t)80 * 640 * 2);
  half_t* whk = (half_t*)alloc((size_t)80 * 640 * 2);
  half_t* whv = (half_t*)alloc((size_t)640 * 640 * 2);

  float* out_ls = (float*)d_out;        // (8,2)
  float* l3feat = (float*)d_out + 16;   // (8,1024)

  auto cdiv = [](size_t a, size_t b) { return (unsigned)((a + b - 1) / b); };
  auto wconv = [&](const float* W, half_t* Wh, int Cout, int Cin, int Kp) {
    size_t tot = (size_t)Cout * Kp;
    k_wconv<<<cdiv(tot, 256), 256, 0, stream>>>(W, Wh, Cout, Cin, Kp);
  };
  auto gemm = [&](const half_t* X, const half_t* W, const float* bias, float* Yo,
                  int R, int Kp, int Nc) {
    if (Nc % 64 == 0) {
      // branch-free 4-tile kernel: 8 waves/block, each 16 rows x 64 cols
      dim3 g((unsigned)(Nc / 64), (unsigned)(R / 128));
      k_gemm_wmma4<<<g, 256, 0, stream>>>(X, W, bias, Yo, R, Kp, Nc);
    } else {
      // single-tile kernel for Nc=80 (q/k projections — tiny)
      dim3 g((unsigned)(Nc / 16), (unsigned)(R / 16));
      k_gemm_wmma1<<<g, 32, 0, stream>>>(X, W, bias, Yo, R, Kp, Nc);
    }
  };
  auto bnlayer = [&](float* Yo, const float* g, const float* be, half_t* outh, int R, int Nc) {
    k_colstats<<<Nc, 256, 0, stream>>>(Yo, R, Nc, sums, sqs);
    size_t tot = (size_t)R * Nc;
    k_bn_relu<<<cdiv(tot, 256), 256, 0, stream>>>(Yo, sums, sqs, g, be, outh, R, Nc);
  };

  // ---- weight conversions (f32 -> padded f16) ----
  wconv(sa1W[0], w10, 64, 3, 32);   wconv(sa1W[1], w11, 64, 64, 64);  wconv(sa1W[2], w12, 320, 64, 64);
  wconv(sa2W[0], w20, 128, 323, 352); wconv(sa2W[1], w21, 128, 128, 128); wconv(sa2W[2], w22, 640, 128, 128);
  wconv(sa3W[0], w30, 256, 643, 672); wconv(sa3W[1], w31, 512, 256, 256); wconv(sa3W[2], w32, 1024, 512, 512);
  wconv(qW, whq, 80, 640, 640); wconv(kW, whk, 80, 640, 640); wconv(vW, whv, 640, 640, 640);

  // ---- stage 0: transpose ----
  k_transpose_xyz<<<cdiv((size_t)Bv * N1 * 3, 256), 256, 0, stream>>>(xyz, pts1, Bv, N1);

  // ---- SA1 ----
  k_fps<<<Bv, 256, 0, stream>>>(pts1, N1, M1, cent1);
  k_gather_xyz<<<cdiv((size_t)Bv * M1, 256), 256, 0, stream>>>(pts1, cent1, newxyz1, Bv, N1, M1);
  k_ball_query<<<cdiv((size_t)Bv * M1, 128), 128, 0, stream>>>(pts1, newxyz1, Bv, N1, M1, S1,
                                                               0.2f * 0.2f, idx1);
  k_group_xyz<<<cdiv((size_t)R1, 256), 256, 0, stream>>>(pts1, newxyz1, idx1, HA, Bv, N1, M1, S1, 32);
  gemm(HA, w10, sa1b[0], Y, R1, 32, 64);   bnlayer(Y, sa1g[0], sa1be[0], HB, R1, 64);
  gemm(HB, w11, sa1b[1], Y, R1, 64, 64);   bnlayer(Y, sa1g[1], sa1be[1], HA, R1, 64);
  gemm(HA, w12, sa1b[2], Y, R1, 64, 320);  bnlayer(Y, sa1g[2], sa1be[2], nullptr, R1, 320);
  k_maxpool<<<cdiv((size_t)Bv * M1 * 320, 256), 256, 0, stream>>>(Y, l1_p, Bv * M1, S1, 320);

  // ---- SA2 ----
  k_fps<<<Bv, 256, 0, stream>>>(newxyz1, M1, M2, cent2);
  k_gather_xyz<<<cdiv((size_t)Bv * M2, 256), 256, 0, stream>>>(newxyz1, cent2, newxyz2, Bv, M1, M2);
  k_ball_query<<<cdiv((size_t)Bv * M2, 128), 128, 0, stream>>>(newxyz1, newxyz2, Bv, M1, M2, S2,
                                                               0.4f * 0.4f, idx2);
  k_group_feat<<<cdiv((size_t)R2, 128), 128, 0, stream>>>(newxyz1, newxyz2, idx2, l1_p, HA,
                                                          Bv, M1, M2, S2, 320, 352);
  gemm(HA, w20, sa2b[0], Y, R2, 352, 128); bnlayer(Y, sa2g[0], sa2be[0], HB, R2, 128);
  gemm(HB, w21, sa2b[1], Y, R2, 128, 128); bnlayer(Y, sa2g[1], sa2be[1], HA, R2, 128);
  gemm(HA, w22, sa2b[2], Y, R2, 128, 640); bnlayer(Y, sa2g[2], sa2be[2], nullptr, R2, 640);
  k_maxpool<<<cdiv((size_t)Bv * M2 * 640, 256), 256, 0, stream>>>(Y, l2_p, Bv * M2, S2, 640);

  // ---- self-attention on l2_p (8,128,640) ----
  k_f2h<<<cdiv((size_t)R3 * 640, 256), 256, 0, stream>>>(l2_p, HB, (size_t)R3 * 640);
  gemm(HB, whq, qb, qf, R3, 640, 80);
  gemm(HB, whk, kb, kf, R3, 640, 80);
  gemm(HB, whv, vb, vf, R3, 640, 640);
  k_attscore<<<cdiv((size_t)Bv * P * P, 256), 256, 0, stream>>>(qf, kf, att, Bv, P, 80);
  k_softmax_rows<<<cdiv((size_t)Bv * P, 128), 128, 0, stream>>>(att, Bv * P, P);
  k_attout<<<cdiv((size_t)Bv * P * 640, 256), 256, 0, stream>>>(att, vf, l2_p, gamma, l2att,
                                                                Bv, P, 640);
  k_maxpool<<<cdiv((size_t)Bv * 640, 256), 256, 0, stream>>>(l2att, l2feat, Bv, P, 640);

  // ---- SA3 (group_all) ----
  k_group_all<<<cdiv((size_t)R3, 128), 128, 0, stream>>>(newxyz2, l2att, HA, Bv, P, 640, 672);
  gemm(HA, w30, sa3b[0], Y, R3, 672, 256);  bnlayer(Y, sa3g[0], sa3be[0], HB, R3, 256);
  gemm(HB, w31, sa3b[1], Y, R3, 256, 512);  bnlayer(Y, sa3g[1], sa3be[1], HA, R3, 512);
  gemm(HA, w32, sa3b[2], Y, R3, 512, 1024); bnlayer(Y, sa3g[2], sa3be[2], nullptr, R3, 1024);
  k_maxpool<<<cdiv((size_t)Bv * 1024, 256), 256, 0, stream>>>(Y, l3feat, Bv, P, 1024);

  // ---- channel attention + head ----
  k_concat_feat<<<cdiv((size_t)Bv * 1664, 256), 256, 0, stream>>>(l3feat, l2feat, xcat, Bv);
  k_linear<<<cdiv((size_t)Bv * 104, 128), 128, 0, stream>>>(xcat, caW1, nullptr, a1, Bv, 1664, 104, 1);
  k_linear<<<cdiv((size_t)Bv * 1664, 128), 128, 0, stream>>>(a1, caW2, nullptr, a2, Bv, 104, 1664, 0);
  k_sig_res<<<cdiv((size_t)Bv * 1664, 256), 256, 0, stream>>>(xcat, a2, Bv * 1664);

  k_linear<<<cdiv((size_t)Bv * 512, 128), 128, 0, stream>>>(xcat, f1W, f1b, h1, Bv, 1664, 512, 0);
  k_bn1d_relu<<<cdiv(512, 128), 128, 0, stream>>>(h1, b1g, b1b, Bv, 512);
  k_linear<<<cdiv((size_t)Bv * 256, 128), 128, 0, stream>>>(h1, f2W, f2b, h2, Bv, 512, 256, 0);
  k_bn1d_relu<<<cdiv(256, 128), 128, 0, stream>>>(h2, b2g, b2b, Bv, 256);
  k_linear<<<cdiv((size_t)Bv * 2, 64), 64, 0, stream>>>(h2, f3W, f3b, logits, Bv, 256, 2, 0);
  k_logsoftmax2<<<1, 64, 0, stream>>>(logits, out_ls, Bv);
}
